// SpaceFeatureAttention_49538152792902
// MI455X (gfx1250) — matrix operand order repaired and checked by reading, not compiled, not run on hardware
//
#include <hip/hip_runtime.h>
#include <math.h>

// Problem constants
#define BATCH 16
#define CINCH 64
#define COUTC 64
#define SDIM  128
#define TDIM  512
#define DDIM  (COUTC * TDIM)   // 32768
#define NTOT  (SDIM * TDIM)    // 65536 columns per batch for the conv GEMM
#define KSPLIT 32
#define KCHUNK (DDIM / KSPLIT) // 1024

typedef __bf16 bhalf;
typedef __attribute__((ext_vector_type(16))) __bf16 v16bf;
typedef __attribute__((ext_vector_type(8)))  float  v8f;
typedef __attribute__((ext_vector_type(4)))  unsigned v4u;
typedef __attribute__((ext_vector_type(8)))  int      v8i;
typedef __attribute__((ext_vector_type(4)))  int      v4i;

// Address-space-qualified vector pointers for the async-DMA builtin.
typedef __attribute__((address_space(1))) v4i gv4i;
typedef __attribute__((address_space(3))) v4i lv4i;

#if __has_builtin(__builtin_amdgcn_global_load_async_to_lds_b128)
#define HAVE_ASYNC_LDS 1
#endif
#if __has_builtin(__builtin_amdgcn_tensor_load_to_lds)
#define HAVE_TDM 1
#endif

// ---------------------------------------------------------------- helpers ---

// 16-byte global -> LDS transfer. Async DMA (ASYNCcnt) when available,
// synchronous register bounce otherwise.
static __device__ inline void g2l_b128(void* lptr, const void* gptr) {
#if defined(HAVE_ASYNC_LDS)
  __builtin_amdgcn_global_load_async_to_lds_b128((gv4i*)gptr, (lv4i*)lptr, 0,
                                                 0);
#else
  __builtin_memcpy(lptr, gptr, 16);
#endif
}

static __device__ inline void wait_async0() {
#if defined(HAVE_ASYNC_LDS)
#if __has_builtin(__builtin_amdgcn_s_wait_asynccnt)
  __builtin_amdgcn_s_wait_asynccnt(0);
#else
  asm volatile("s_wait_asynccnt 0" ::: "memory");
#endif
#endif
}

// 32-byte (16 x bf16) LDS load, alignment-agnostic.
static __device__ inline v16bf lds_ld16(const bhalf* p) {
  v16bf r;
  __builtin_memcpy(&r, p, sizeof(r));
  return r;
}

// A-matrix fragment (16x32 bf16) from a bf16 row-major LDS matrix.
// Layout: lanes 0-15 (M=lane) hold K = {kb+0..15},
//         lanes 16-31 (M=lane-16) hold K = {kb+16..31}.
static __device__ inline v16bf a_frag_bf(const bhalf* W, int ld, int mbase,
                                         int kbase, int lane) {
  const int m    = mbase + (lane & 15);
  const int half = (lane >> 4) & 1;
  const bhalf* row = W + m * ld + kbase + half * 16;
  v16bf r;
#pragma unroll
  for (int j = 0; j < 8; ++j) r[j] = row[j];
#pragma unroll
  for (int j = 0; j < 8; ++j) r[8 + j] = row[8 + j];
  return r;
}

static __device__ inline v8f wmma_bf16(v16bf a, v16bf b, v8f c) {
  return __builtin_amdgcn_wmma_f32_16x16x32_bf16(
      /*neg_a=*/false, a, /*neg_b=*/false, b,
      /*c_mod=*/(short)0, c, /*reuse_a=*/false, /*reuse_b=*/false);
}

// ----------------------------------------------------------- weff kernel ---
// weff = w14 @ w13 (64x64), beff = w14 @ b13. Folds the trailing 1x1 conv
// into the v-projection (conv and P@v commute): removes a 256MB round-trip.
__global__ void weff_kernel(const float* __restrict__ w13,
                            const float* __restrict__ b13,
                            const float* __restrict__ w14,
                            float* __restrict__ weff,
                            float* __restrict__ beff) {
  const int i = threadIdx.x;  // 64 threads
  for (int cp = 0; cp < CINCH; ++cp) {
    float acc = 0.f;
    for (int c = 0; c < COUTC; ++c) acc += w14[i * COUTC + c] * w13[c * CINCH + cp];
    weff[i * CINCH + cp] = acc;
  }
  float bb = 0.f;
  for (int c = 0; c < COUTC; ++c) bb += w14[i * COUTC + c] * b13[c];
  beff[i] = bb;
}

// ----------------------------------------------------------- conv kernel ---
// Fused q/k/v' projections. Weights are converted to bf16 in LDS ONCE
// (so WMMA loops do pure ds_load_b128, no cvt chains), then the block loops
// over 8 column-chunks of 256 to amortize the staging cost.
__global__ __launch_bounds__(256) void conv3_kernel(
    const float* __restrict__ x,
    const float* __restrict__ w11, const float* __restrict__ b11,
    const float* __restrict__ w12, const float* __restrict__ b12,
    const float* __restrict__ weff, const float* __restrict__ beff,
    bhalf* __restrict__ q, bhalf* __restrict__ k, bhalf* __restrict__ vp) {
  __shared__ bhalf xs[256 * 72];        // [n_local][c], padded stride 72
  __shared__ bhalf wlds[3 * 64 * 64];   // w11 | w12 | weff, bf16
  __shared__ float blds[3 * 64];        // b11 | b12 | beff

  const int tid  = threadIdx.x;
  const int lane = tid & 31;
  const int wave = tid >> 5;
  const int b    = blockIdx.y;

  // Weights -> LDS, converted to bf16 once.
  for (int i = tid; i < 3 * 4096; i += 256) {
    const int widx = i >> 12, r = i & 4095;
    const float v = (widx == 0) ? w11[r] : (widx == 1) ? w12[r] : weff[r];
    wlds[i] = (bhalf)v;
  }
  if (tid < 192) {
    const int widx = tid >> 6, r = tid & 63;
    blds[tid] = (widx == 0) ? b11[r] : (widx == 1) ? b12[r] : beff[r];
  }

  const int nb   = wave * 32;  // this wave's 32 columns within the chunk
  const int half = (lane >> 4) & 1;
  bhalf* const outs[3] = {q, k, vp};

  for (int ch = 0; ch < 8; ++ch) {
    const int n0 = (blockIdx.x * 8 + ch) * 256;
    if (ch) __syncthreads();  // previous chunk fully consumed before restage
    // x tile -> LDS (coalesced per c-row; cvt to bf16; store transposed)
    {
      const size_t xbase = (size_t)b * CINCH * NTOT + n0 + tid;
#pragma unroll 4
      for (int c = 0; c < CINCH; ++c)
        xs[tid * 72 + c] = (bhalf)x[xbase + (size_t)c * NTOT];
    }
    __syncthreads();

    // B fragments shared by all three projections: [kc][nt]
    v16bf bfr[2][2];
#pragma unroll
    for (int kc = 0; kc < 2; ++kc)
#pragma unroll
      for (int nt = 0; nt < 2; ++nt) {
        const int nl = nb + nt * 16 + (lane & 15);
        bfr[kc][nt] = lds_ld16(&xs[nl * 72 + kc * 32 + half * 16]);
      }

#pragma unroll
    for (int widx = 0; widx < 3; ++widx) {
      v8f acc[4][2] = {};
#pragma unroll
      for (int mt = 0; mt < 4; ++mt)
#pragma unroll
        for (int kc = 0; kc < 2; ++kc) {
          const v16bf a =
              a_frag_bf(&wlds[widx * 4096], 64, mt * 16, kc * 32, lane);
#pragma unroll
          for (int nt = 0; nt < 2; ++nt)
            acc[mt][nt] = wmma_bf16(a, bfr[kc][nt], acc[mt][nt]);
        }
      // Epilogue: +bias, cvt bf16, scatter to [b, s, c*T + t]
      bhalf* dst = outs[widx];
#pragma unroll
      for (int mt = 0; mt < 4; ++mt)
#pragma unroll
        for (int nt = 0; nt < 2; ++nt) {
          const int nl = nb + nt * 16 + (lane & 15);
          const int n  = n0 + nl;
          const int s  = n >> 9;   // n / T
          const int t  = n & 511;  // n % T
          const size_t base = ((size_t)b * SDIM + s) * DDIM + t;
#pragma unroll
          for (int r = 0; r < 8; ++r) {
            const int m = mt * 16 + r + half * 8;
            const float v = acc[mt][nt][r] + blds[widx * 64 + m];
            dst[base + (size_t)m * TDIM] = (bhalf)v;
          }
        }
    }
  }
}

// ------------------------------------------------------ QK^T (split-K) -----
// attn_partial[b][kch][s1][s2] over 1024 d's, double-buffered LDS staging
// via async global->LDS DMA (ASYNCcnt) overlapping DMA of step n+1 with
// WMMA of step n. Deterministic (no atomics).
__global__ __launch_bounds__(256) void attn_qk_kernel(
    const bhalf* __restrict__ q, const bhalf* __restrict__ k,
    float* __restrict__ attnp) {
  __shared__ bhalf qs[2][128 * 40];  // [buf][s1][kk], padded
  __shared__ bhalf ks[2][128 * 40];  // [buf][s2][kk], padded

  const int tid  = threadIdx.x;
  const int lane = tid & 31;
  const int wave = tid >> 5;
  const int b    = blockIdx.y;
  const int d0   = blockIdx.x * KCHUNK;

  const int mb   = (wave & 3) * 32;   // 4 waves across M
  const int nb   = (wave >> 2) * 64;  // 2 waves across N
  const int half = (lane >> 4) & 1;

  // Per-thread staging coordinates (16B q + 16B k per iter, 2 iters)
  const int lin0 = tid * 8;
  const int s_a = lin0 >> 5, kk_a = lin0 & 31;
  const int lin1 = 2048 + tid * 8;
  const int s_b = lin1 >> 5, kk_b = lin1 & 31;

#define STAGE_QK(buf, dk)                                                  \
  do {                                                                     \
    const size_t ga = ((size_t)b * SDIM + s_a) * DDIM + (dk) + kk_a;       \
    const size_t gb = ((size_t)b * SDIM + s_b) * DDIM + (dk) + kk_b;       \
    g2l_b128(&qs[buf][s_a * 40 + kk_a], &q[ga]);                           \
    g2l_b128(&ks[buf][s_a * 40 + kk_a], &k[ga]);                           \
    g2l_b128(&qs[buf][s_b * 40 + kk_b], &q[gb]);                           \
    g2l_b128(&ks[buf][s_b * 40 + kk_b], &k[gb]);                           \
  } while (0)

  v8f acc[2][4] = {};

  STAGE_QK(0, d0);
  for (int step = 0; step < KCHUNK / 32; ++step) {
    const int cur = step & 1;
    wait_async0();     // my async loads for buffer `cur` have landed
    __syncthreads();   // everyone's loads visible; prev compute done
    if (step + 1 < KCHUNK / 32) STAGE_QK(cur ^ 1, d0 + (step + 1) * 32);

    const v16bf a0 = a_frag_bf(qs[cur], 40, mb, 0, lane);
    const v16bf a1 = a_frag_bf(qs[cur], 40, mb + 16, 0, lane);
#pragma unroll
    for (int nt = 0; nt < 4; ++nt) {
      const int nl = nb + nt * 16 + (lane & 15);
      const v16bf bf = lds_ld16(&ks[cur][nl * 40 + half * 16]);
      acc[0][nt] = wmma_bf16(a0, bf, acc[0][nt]);
      acc[1][nt] = wmma_bf16(a1, bf, acc[1][nt]);
    }
  }
#undef STAGE_QK

#pragma unroll
  for (int mt = 0; mt < 2; ++mt)
#pragma unroll
    for (int nt = 0; nt < 4; ++nt) {
      const int s2 = nb + nt * 16 + (lane & 15);
#pragma unroll
      for (int r = 0; r < 8; ++r) {
        const int s1 = mb + mt * 16 + r + half * 8;
        attnp[(((size_t)b * KSPLIT + blockIdx.x) * SDIM + s1) * SDIM + s2] =
            acc[mt][nt][r];
      }
    }
}

// -------------------------------------------------------------- softmax ----
__global__ __launch_bounds__(128) void softmax_kernel(
    const float* __restrict__ attnp, bhalf* __restrict__ P) {
  const int row = blockIdx.x;
  const int b = row >> 7, s1 = row & 127;
  const int j = threadIdx.x;

  float acc = 0.f;
  for (int kc = 0; kc < KSPLIT; ++kc)
    acc += attnp[(((size_t)b * KSPLIT + kc) * SDIM + s1) * SDIM + j];
  const float kScaleInv = 1.0f / 181.01933598375618f;  // 1/sqrt(64*512)
  const float val = acc * kScaleInv;

  __shared__ float red[128];
  red[j] = val;
  __syncthreads();
  for (int off = 64; off > 0; off >>= 1) {
    if (j < off) red[j] = fmaxf(red[j], red[j + off]);
    __syncthreads();
  }
  const float m = red[0];
  __syncthreads();
  const float e = expf(val - m);
  red[j] = e;
  __syncthreads();
  for (int off = 64; off > 0; off >>= 1) {
    if (j < off) red[j] += red[j + off];
    __syncthreads();
  }
  const float ssum = red[0];
  P[((size_t)b * SDIM + s1) * SDIM + j] = (bhalf)(e / (1.0f + ssum));
}

// ----------------------------------------------------- P@v' + epilogue -----
// out[b,i,s1,t] = alpha[i,s1,t]*(sum_s2 P[s1,s2] v'[s2,i*T+t] + b14[i]) + x.
// P (32KB contiguous) is staged by the Tensor Data Mover when available.
__global__ __launch_bounds__(256) void attn_pv_kernel(
    const bhalf* __restrict__ P, const bhalf* __restrict__ vp,
    const float* __restrict__ b14, const float* __restrict__ alpha,
    const float* __restrict__ x, float* __restrict__ out) {
  __shared__ bhalf Ps[128 * 128];  // [s1][s2]
  __shared__ bhalf Vt[128 * 136];  // [n_local][s2] (transposed), padded

  const int tid  = threadIdx.x;
  const int lane = tid & 31;
  const int wave = tid >> 5;
  const int b    = blockIdx.y;
  const int n0   = blockIdx.x * 128;

#if defined(HAVE_TDM)
  if (wave == 0) {
    // Tensor DMA descriptor: 1-D tile of 16384 bf16 (32KB) -> LDS.
    const unsigned lds_off =
        (unsigned)(unsigned long long)(__attribute__((address_space(3)))
                                           const void*)&Ps[0];
    const unsigned long long ga =
        (unsigned long long)(const void*)(P + (size_t)b * SDIM * SDIM);
    const v4u g0 = {1u,  // count=1 (valid descriptor)
                    lds_off, (unsigned)(ga & 0xffffffffu),
                    (unsigned)((ga >> 32) & 0x1ffffffu) | (2u << 30)};  // type=2
    const v8i g1 = {(int)(1u << 16),      // data_size = 2 bytes
                    (int)(16384u << 16),  // tensor_dim0 = 16384 (bits 79:48)
                    (int)(1u << 16),      // tensor_dim1 = 1    (bits 111:80)
                    (int)(16384u << 16),  // tile_dim0   = 16384 (bits 127:112)
                    0,                    // tile_dim1/2 unused
                    16384, 0, 0};         // tensor_dim0_stride = 16384
    const v4i g2 = {0, 0, 0, 0};
    const v4i g3 = {0, 0, 0, 0};
    const v8i g4 = {0, 0, 0, 0, 0, 0, 0, 0};  // unused (6-arg toolchain form)
    __builtin_amdgcn_tensor_load_to_lds(g0, g1, g2, g3, g4, 0);
#if __has_builtin(__builtin_amdgcn_s_wait_tensorcnt)
    __builtin_amdgcn_s_wait_tensorcnt((short)0);
#else
    asm volatile("s_wait_tensorcnt 0" ::: "memory");
#endif
  }
#else
#pragma unroll
  for (int it = 0; it < 8; ++it) {
    const int lin = it * 2048 + tid * 8;
    __builtin_memcpy(&Ps[lin], &P[(size_t)b * SDIM * SDIM + lin], 16);
  }
#endif

  // Stage v' chunk transposed: coalesced global read, scattered LDS write
#pragma unroll
  for (int it = 0; it < 8; ++it) {
    const int lin = it * 2048 + tid * 8;
    const int s2 = lin >> 7, nl0 = lin & 127;
    bhalf tmp[8];
    __builtin_memcpy(tmp, &vp[((size_t)b * SDIM + s2) * DDIM + n0 + nl0], 16);
#pragma unroll
    for (int e = 0; e < 8; ++e) Vt[(nl0 + e) * 136 + s2] = tmp[e];
  }
  __syncthreads();

  const int mb   = (wave & 3) * 32;
  const int nb   = (wave >> 2) * 64;
  const int half = (lane >> 4) & 1;

  v8f acc[2][4] = {};
#pragma unroll
  for (int kc = 0; kc < 4; ++kc) {
    const v16bf a0 = a_frag_bf(Ps, 128, mb, kc * 32, lane);
    const v16bf a1 = a_frag_bf(Ps, 128, mb + 16, kc * 32, lane);
#pragma unroll
    for (int nt = 0; nt < 4; ++nt) {
      const int nl = nb + nt * 16 + (lane & 15);
      const v16bf bf = lds_ld16(&Vt[nl * 136 + kc * 32 + half * 16]);
      acc[0][nt] = wmma_bf16(a0, bf, acc[0][nt]);
      acc[1][nt] = wmma_bf16(a1, bf, acc[1][nt]);
    }
  }

#pragma unroll
  for (int mt = 0; mt < 2; ++mt)
#pragma unroll
    for (int nt = 0; nt < 4; ++nt) {
      const int nl = nb + nt * 16 + (lane & 15);
      const int n  = n0 + nl;
      const int ic = n >> 9;   // channel i
      const int t  = n & 511;  // time t
      const float bias = b14[ic];
#pragma unroll
      for (int r = 0; r < 8; ++r) {
        const int s1 = mb + mt * 16 + r + half * 8;
        const size_t off  = (((size_t)b * COUTC + ic) * SDIM + s1) * TDIM + t;
        const size_t aoff = ((size_t)ic * SDIM + s1) * TDIM + t;
        const float y = acc[mt][nt][r] + bias;
        out[off] = alpha[aoff] * y + x[off];
      }
    }
}

// ---------------------------------------------------------------- launch ---
extern "C" void kernel_launch(void* const* d_in, const int* in_sizes, int n_in,
                              void* d_out, int out_size, void* d_ws,
                              size_t ws_size, hipStream_t stream) {
  const float* x     = (const float*)d_in[0];
  const float* w11   = (const float*)d_in[1];
  const float* b11   = (const float*)d_in[2];
  const float* w12   = (const float*)d_in[3];
  const float* b12   = (const float*)d_in[4];
  const float* w13   = (const float*)d_in[5];
  const float* b13   = (const float*)d_in[6];
  const float* w14   = (const float*)d_in[7];
  const float* b14   = (const float*)d_in[8];
  const float* alpha = (const float*)d_in[9];
  float* out = (float*)d_out;

  // Workspace carve-up
  const size_t QE = (size_t)BATCH * SDIM * DDIM;  // 67,108,864 elems
  bhalf* q  = (bhalf*)d_ws;
  bhalf* k  = q + QE;
  bhalf* vp = k + QE;
  float* attnp = (float*)(vp + QE);
  const size_t AE = (size_t)BATCH * KSPLIT * SDIM * SDIM;  // fp32 partials
  bhalf* P = (bhalf*)(attnp + AE);
  float* weff = (float*)(P + (size_t)BATCH * SDIM * SDIM);
  float* beff = weff + COUTC * CINCH;

  weff_kernel<<<dim3(1), dim3(64), 0, stream>>>(w13, b13, w14, weff, beff);
  conv3_kernel<<<dim3(NTOT / (256 * 8), BATCH), dim3(256), 0, stream>>>(
      x, w11, b11, w12, b12, weff, beff, q, k, vp);
  attn_qk_kernel<<<dim3(KSPLIT, BATCH), dim3(256), 0, stream>>>(q, k, attnp);
  softmax_kernel<<<dim3(BATCH * SDIM), dim3(128), 0, stream>>>(attnp, P);
  attn_pv_kernel<<<dim3(DDIM / 128, BATCH), dim3(256), 0, stream>>>(
      P, vp, b14, alpha, x, out);
}
